// DenseGATLayer_20066087206934
// MI455X (gfx1250) — matrix operand order repaired and checked by reading, not compiled, not run on hardware
//
#include <hip/hip_runtime.h>
#include <hip/hip_bf16.h>

// ---------------------------------------------------------------------------
// Dense GAT layer for MI455X (gfx1250, wave32, WMMA).
// B=2, N=4096, IN_DIM=256, H=4, D=64.
//
// Roofline: adj (134MB int32) -> read once, ~5.8us @ 23.3TB/s. Aggregation
// (17.2 GFLOP + 25% for WMMA row-sums) on v_wmma_f32_16x16x32_bf16. h (4MB)
// and all small tensors are L2-resident (192MB); attention reads fragments
// directly from global with b128 loads (no LDS, no barriers).
//
// Softmax separability: with m_i = leaky(s_i + max_j d_j) (valid bound, leaky
// monotone) and leaky(z) = max(z, 0.2z):
//   exp(leaky(s_i+d_j) - m_i) = max( e^{s_i-m_i} * e^{d_j},
//                                    e^{0.2 s_i-m_i} * e^{0.2 d_j} )
// so ALL 134M inner-loop exps become 2x32K precomputed tables + 2 exps/row:
// hot loop is 2 v_mul + v_max + v_cndmask per p, zero transcendentals.
// Row sums come from an extra WMMA against a constant all-ones B matrix
// (every C column = row sum, already in the right lane), so no shuffles.
// ---------------------------------------------------------------------------

typedef __attribute__((ext_vector_type(16))) __bf16 v16bf;
typedef __attribute__((ext_vector_type(8)))  float  v8f;

#define NN   4096
#define BB   2
#define HHN  4
#define DDN  64
#define KDIM 256            // IN_DIM == H*D
#define MTOT (BB * NN)      // 8192 flat rows

union ABf   { v16bf v; __bf16 e[16]; };
union Bfrag { v16bf v; uint4 q[2]; __bf16 e[16]; };
union CF    { v8f   v; float  e[8];  };
union I4u   { int4   q; int   e[4]; };
union F4u   { float4 q; float e[4]; };
union HPack { uint4  q; __bf16 e[8]; };

// A-matrix (16x32 bf16) K offset for element pair v (0..7), lane half:
//   lanes 0-15: VGPR0-3 -> K=0..7,  VGPR4-7 -> K=16..23
//   lanes16-31: VGPR0-3 -> K=8..15, VGPR4-7 -> K=24..31
__device__ __forceinline__ int a_koff(int v, int half) {
    return 2 * (v & 3) + ((v >> 2) << 4) + 8 * half;
}
// Contiguous-k offset, unused (B loads use a_koff): lanes 0-15 hold K=0..15 (2/VGPR), lanes 16-31 K=16..31.
__device__ __forceinline__ int b_koff(int v, int half) {
    return 2 * v + 16 * half;
}

// ---------------------------------------------------------------------------
// Kernel 1: projection  hT[o, m] = sum_i x[m,i] * W[o,i]  (transposed output).
// M=8192, K=256, O=256. Block: 128 threads / 4 waves; tile 16 rows x 64 cols.
// Transposed C store: 8 consecutive rows of one column -> ONE b128 store/lane.
// ---------------------------------------------------------------------------
__global__ __launch_bounds__(128)
void gat_proj_kernel(const float* __restrict__ x, const float* __restrict__ Wp,
                     __bf16* __restrict__ hT) {
    __shared__ __bf16 xA[16][264];   // padded vs bank conflicts
    __shared__ __bf16 Wb[64][264];
    const int t    = threadIdx.x;
    const int lane = t & 31, wave = t >> 5;
    const int i0   = blockIdx.x * 16;
    const int o0   = blockIdx.y * 64;

    #pragma unroll
    for (int q = 0; q < 8; ++q) {       // x tile: 16x256 f32 = 1024 float4
        int idx = q * 128 + t;
        int row = idx >> 6, col4 = idx & 63;
        float4 vv = ((const float4*)x)[(size_t)(i0 + row) * 64 + col4];
        int c = col4 << 2;
        xA[row][c + 0] = (__bf16)vv.x; xA[row][c + 1] = (__bf16)vv.y;
        xA[row][c + 2] = (__bf16)vv.z; xA[row][c + 3] = (__bf16)vv.w;
    }
    #pragma unroll
    for (int q = 0; q < 32; ++q) {      // W tile: 64x256 f32 = 4096 float4
        int idx = q * 128 + t;
        int row = idx >> 6, col4 = idx & 63;
        float4 vv = ((const float4*)Wp)[(size_t)(o0 + row) * 64 + col4];
        int c = col4 << 2;
        Wb[row][c + 0] = (__bf16)vv.x; Wb[row][c + 1] = (__bf16)vv.y;
        Wb[row][c + 2] = (__bf16)vv.z; Wb[row][c + 3] = (__bf16)vv.w;
    }
    __syncthreads();

    const int m = lane & 15, half = lane >> 4, n = lane & 15;
    CF acc;
    #pragma unroll
    for (int r = 0; r < 8; ++r) acc.e[r] = 0.f;

    #pragma unroll
    for (int kk = 0; kk < KDIM; kk += 32) {
        ABf a, b;
        #pragma unroll
        for (int v = 0; v < 8; ++v) {
            int ka = kk + a_koff(v, half);
            a.e[2 * v]     = xA[m][ka];
            a.e[2 * v + 1] = xA[m][ka + 1];
            int kb = kk + a_koff(v, half);
            b.e[2 * v]     = Wb[wave * 16 + n][kb];
            b.e[2 * v + 1] = Wb[wave * 16 + n][kb + 1];
        }
        acc.v = __builtin_amdgcn_wmma_f32_16x16x32_bf16(
            false, a.v, false, b.v, (short)0, acc.v, false, false);
    }

    HPack hp;
    #pragma unroll
    for (int r = 0; r < 8; ++r) hp.e[r] = (__bf16)acc.e[r];
    int go = o0 + wave * 16 + n;
    *(uint4*)&hT[(size_t)go * MTOT + i0 + 8 * half] = hp.q;
}

// ---------------------------------------------------------------------------
// Kernel 2: srcT/dstT dots + separable-softmax tables e^{d}, e^{0.2 d}.
// Fully coalesced over hT's transposed layout.
// ---------------------------------------------------------------------------
__global__ __launch_bounds__(256)
void gat_srcdst_kernel(const __bf16* __restrict__ hT,
                       const float* __restrict__ aS, const float* __restrict__ aD,
                       float* __restrict__ srcT, float* __restrict__ dstT,
                       float* __restrict__ ev1T, float* __restrict__ ev2T) {
    int tg  = blockIdx.x * 256 + threadIdx.x;   // 0 .. H*MTOT-1
    int row = tg & (MTOT - 1);
    int hh  = tg >> 13;                          // MTOT = 2^13
    float s = 0.f, d = 0.f;
    #pragma unroll 8
    for (int dd = 0; dd < DDN; ++dd) {
        float hv = (float)hT[(size_t)(hh * DDN + dd) * MTOT + row];
        s += hv * aS[hh * DDN + dd];
        d += hv * aD[hh * DDN + dd];
    }
    srcT[hh * MTOT + row] = s;
    dstT[hh * MTOT + row] = d;
    ev1T[hh * MTOT + row] = __expf(d);
    ev2T[hh * MTOT + row] = __expf(0.2f * d);
}

// ---------------------------------------------------------------------------
// Kernel 3: maxW[b,h] = max_j dstT[h, b*N+j].  8 blocks of 256, coalesced.
// ---------------------------------------------------------------------------
__global__ __launch_bounds__(256)
void gat_maxdst_kernel(const float* __restrict__ dstT, float* __restrict__ maxW) {
    __shared__ float red[256];
    int bh = blockIdx.x, b = bh >> 2, hh = bh & 3;
    float mx = -1e30f;
    for (int j = threadIdx.x; j < NN; j += 256)
        mx = fmaxf(mx, dstT[(size_t)hh * MTOT + b * NN + j]);
    red[threadIdx.x] = mx;
    __syncthreads();
    for (int s2 = 128; s2 > 0; s2 >>= 1) {
        if (threadIdx.x < (unsigned)s2)
            red[threadIdx.x] = fmaxf(red[threadIdx.x], red[threadIdx.x + s2]);
        __syncthreads();
    }
    if (threadIdx.x == 0) maxW[bh] = red[0];
}

// ---------------------------------------------------------------------------
// Kernel 4: fused masked softmax + aggregation. LDS-free, barrier-free,
// transcendental-free inner loop.
// Block = (b, 32-row i-chunk); 4 waves = 4 heads share adj lines via WGP$.
// Each wave handles TWO 16-row i-tiles (register reuse of B fragments halves
// h's L2 read volume). Per 32-wide j-tile:
//   adj  : 2 x 4 aligned int4 loads (per i-tile)
//   ev1/2: 4+4 aligned float4 loads (shared by both i-tiles)
//   h    : 2 uint4 per 16x16 B-tile (shared by both i-tiles)
//   p    : 2 v_mul + v_max + v_cndmask each; row sums via WMMA vs ones-B.
// ---------------------------------------------------------------------------
__global__ __launch_bounds__(128)
void gat_attn_kernel(const int* __restrict__ adj, const __bf16* __restrict__ hT,
                     const float* __restrict__ srcT,
                     const float* __restrict__ ev1T, const float* __restrict__ ev2T,
                     const float* __restrict__ maxW, float* __restrict__ outF) {
    const int t    = threadIdx.x;
    const int lane = t & 31;
    const int hh   = t >> 5;                 // wave == head
    const int b    = blockIdx.x >> 7;        // 128 i-chunks per batch
    const int i0   = (blockIdx.x & 127) * 32;
    const int m = lane & 15, half = lane >> 4, n = lane & 15;

    const float mdst = maxW[b * HHN + hh];

    float esm1[2], esm2[2];
    const int* adjL[2];
    #pragma unroll
    for (int it = 0; it < 2; ++it) {
        float s = srcT[(size_t)hh * MTOT + b * NN + i0 + it * 16 + m];
        float z = s + mdst;
        float mrow = fmaxf(z, 0.2f * z);          // leaky, monotone bound
        esm1[it] = __expf(s - mrow);
        esm2[it] = __expf(0.2f * s - mrow);
        adjL[it] = adj + ((size_t)(b * NN + i0 + it * 16 + m)) * NN;
    }
    const float*  ev1L = ev1T + (size_t)hh * MTOT + b * NN;
    const float*  ev2L = ev2T + (size_t)hh * MTOT + b * NN;
    const __bf16* hL   = hT   + (size_t)(hh * DDN + n) * MTOT + b * NN;

    Bfrag onesB;                                  // constant all-ones B matrix
    #pragma unroll
    for (int i = 0; i < 16; ++i) onesB.e[i] = (__bf16)1.0f;

    CF acc[2][4], accl[2];
    #pragma unroll
    for (int it = 0; it < 2; ++it) {
        #pragma unroll
        for (int r = 0; r < 8; ++r) accl[it].e[r] = 0.f;
        #pragma unroll
        for (int dt = 0; dt < 4; ++dt)
            #pragma unroll
            for (int r = 0; r < 8; ++r) acc[it][dt].e[r] = 0.f;
    }

    for (int j0 = 0; j0 < NN; j0 += 32) {
        const int jA = j0 + 8 * half;    // A-fragment runs: jA..+7, jA+16..+23
        F4u ev1[4], ev2[4];
        ev1[0] = *(const F4u*)(ev1L + jA);      ev1[1] = *(const F4u*)(ev1L + jA + 4);
        ev1[2] = *(const F4u*)(ev1L + jA + 16); ev1[3] = *(const F4u*)(ev1L + jA + 20);
        ev2[0] = *(const F4u*)(ev2L + jA);      ev2[1] = *(const F4u*)(ev2L + jA + 4);
        ev2[2] = *(const F4u*)(ev2L + jA + 16); ev2[3] = *(const F4u*)(ev2L + jA + 20);

        if (j0 + 32 < NN) {              // global_prefetch_b8 next adj chunk
            __builtin_prefetch(adjL[0] + j0 + 32 + 8 * half, 0, 1);
            __builtin_prefetch(adjL[1] + j0 + 32 + 8 * half, 0, 1);
        }

        ABf a[2];
        #pragma unroll
        for (int it = 0; it < 2; ++it) {
            I4u av[4];
            av[0] = *(const I4u*)(adjL[it] + jA);
            av[1] = *(const I4u*)(adjL[it] + jA + 4);
            av[2] = *(const I4u*)(adjL[it] + jA + 16);
            av[3] = *(const I4u*)(adjL[it] + jA + 20);
            #pragma unroll
            for (int i = 0; i < 16; ++i) {   // e[0..7]=run0, e[8..15]=run1
                int g = i >> 2, sub = i & 3; // constant after unroll
                float t1 = esm1[it] * ev1[g].e[sub];
                float t2 = esm2[it] * ev2[g].e[sub];
                float p  = (av[g].e[sub] > 0) ? fmaxf(t1, t2) : 0.f;
                a[it].e[i] = (__bf16)p;
            }
            accl[it].v = __builtin_amdgcn_wmma_f32_16x16x32_bf16(
                false, a[it].v, false, onesB.v, (short)0, accl[it].v, false, false);
        }

        const int jB = j0 + 8 * half;    // B-fragment runs: jB..+7, jB+16..+23
        #pragma unroll
        for (int dt = 0; dt < 4; ++dt) {
            const __bf16* hp = hL + (size_t)(dt * 16) * MTOT + jB;
            Bfrag bb;
            bb.q[0] = *(const uint4*)(hp);
            bb.q[1] = *(const uint4*)(hp + 16);
            #pragma unroll
            for (int it = 0; it < 2; ++it)
                acc[it][dt].v = __builtin_amdgcn_wmma_f32_16x16x32_bf16(
                    false, a[it].v, false, bb.v, (short)0, acc[it][dt].v, false, false);
        }
    }

    // accl columns all equal the row sum and sit in the matching lane/VGPR.
    #pragma unroll
    for (int it = 0; it < 2; ++it) {
        float rinv[8];
        #pragma unroll
        for (int r = 0; r < 8; ++r)
            rinv[r] = 1.f / fmaxf(accl[it].e[r], 1e-20f);
        #pragma unroll
        for (int dt = 0; dt < 4; ++dt)
            #pragma unroll
            for (int r = 0; r < 8; ++r) {
                int gm = i0 + it * 16 + r + 8 * half;
                __builtin_nontemporal_store(
                    acc[it][dt].e[r] * rinv[r],
                    &outF[((size_t)b * NN + gm) * KDIM + hh * DDN + dt * 16 + n]);
            }
    }
}

// ---------------------------------------------------------------------------
extern "C" void kernel_launch(void* const* d_in, const int* in_sizes, int n_in,
                              void* d_out, int out_size, void* d_ws, size_t ws_size,
                              hipStream_t stream) {
    (void)in_sizes; (void)n_in; (void)out_size; (void)ws_size;
    const float* x   = (const float*)d_in[0];
    const int*   adj = (const int*)d_in[1];
    const float* Wp  = (const float*)d_in[2];
    const float* aS  = (const float*)d_in[3];
    const float* aD  = (const float*)d_in[4];
    float* outF = (float*)d_out;

    char* ws = (char*)d_ws;
    const size_t MB4 = 4u * 1024u * 1024u, T = 131072u;
    __bf16* hT  = (__bf16*)ws;                    // 4 MiB   [256][8192]
    float* srcT = (float*)(ws + MB4);             // 128 KiB [4][8192]
    float* dstT = (float*)(ws + MB4 + 1 * T);     // 128 KiB
    float* ev1T = (float*)(ws + MB4 + 2 * T);     // 128 KiB  e^{d}
    float* ev2T = (float*)(ws + MB4 + 3 * T);     // 128 KiB  e^{0.2 d}
    float* maxW = (float*)(ws + MB4 + 4 * T);     // 32 B

    hipLaunchKernelGGL(gat_proj_kernel,   dim3(512, 4), dim3(128), 0, stream, x, Wp, hT);
    hipLaunchKernelGGL(gat_srcdst_kernel, dim3(128),    dim3(256), 0, stream,
                       hT, aS, aD, srcT, dstT, ev1T, ev2T);
    hipLaunchKernelGGL(gat_maxdst_kernel, dim3(8),      dim3(256), 0, stream, dstT, maxW);
    hipLaunchKernelGGL(gat_attn_kernel,   dim3(256),    dim3(128), 0, stream,
                       adj, hT, srcT, ev1T, ev2T, maxW, outF);
}
